// Attention_62225486184866
// MI455X (gfx1250) — matrix operand order repaired and checked by reading, not compiled, not run on hardware
//
#include <hip/hip_runtime.h>

typedef _Float16 h16;
typedef __attribute__((ext_vector_type(16))) _Float16 v16h;
typedef __attribute__((ext_vector_type(8)))  _Float16 v8h;
typedef __attribute__((ext_vector_type(8)))  float v8f;
typedef __attribute__((ext_vector_type(4)))  float v4f;

#define NW 2048        // windows (B)
#define NT 64          // tokens per window (N)
#define CH 512         // channels (C)
#define NH 16          // heads
#define HD 32          // head dim
#define SCALE 0.17677669529663687f  // 32^-0.5

__device__ __forceinline__ v8f wmma_f16(v16h a, v16h b, v8f c) {
  // D = A(16x32 f16) * B(32x16 f16) + C(16x16 f32)
  return __builtin_amdgcn_wmma_f32_16x16x32_f16(false, a, false, b, (short)0, c,
                                                false, false);
}

// A-operand (16x32 f16, M=rows): lane holds row, two 16B chunks at k=hi*8 and 16+hi*8
__device__ __forceinline__ v16h load_a_frag(const h16* base, int ld, int row,
                                            int k0, int hi) {
  const h16* r = base + (size_t)row * ld + k0;
  v8h a0 = *(const v8h*)(r + hi * 8);
  v8h a1 = *(const v8h*)(r + 16 + hi * 8);
  v16h a;
#pragma unroll
  for (int i = 0; i < 8; ++i) { a[i] = a0[i]; a[i + 8] = a1[i]; }
  return a;
}

// B-operand from [N x K] row-major buffer: lane = column (lo), two 16B chunks at k=hi*8 and 16+hi*8
__device__ __forceinline__ v16h load_b_frag(const h16* base, int ld, int col,
                                            int k0, int hi) {
  const h16* r = base + (size_t)col * ld + k0;
  v8h b0 = *(const v8h*)(r + hi * 8);
  v8h b1 = *(const v8h*)(r + 16 + hi * 8);
  v16h b;
#pragma unroll
  for (int i = 0; i < 8; ++i) { b[i] = b0[i]; b[i + 8] = b1[i]; }
  return b;
}

// ---------------------------------------------------------------- prep ------
__global__ __launch_bounds__(256) void prep_kernel(
    const float* __restrict__ qkv_w, const float* __restrict__ proj_w,
    const float* __restrict__ bias_table, const int* __restrict__ rel_index,
    h16* __restrict__ qkvT, h16* __restrict__ projT, float* __restrict__ biasb) {
  const int stride = gridDim.x * blockDim.x;
  const int t0 = blockIdx.x * blockDim.x + threadIdx.x;
  // qkv_w [512][1536] -> qkvT f16 [1536][512]
  for (int i = t0; i < 1536 * 512; i += stride) {
    int n = i >> 9, k = i & 511;
    qkvT[i] = (h16)qkv_w[k * 1536 + n];
  }
  // proj_w [512][512] -> projT f16 [512][512] (transposed)
  for (int i = t0; i < 512 * 512; i += stride) {
    int n = i >> 9, k = i & 511;
    projT[i] = (h16)proj_w[k * 512 + n];
  }
  // bias [H][64][64] gather
  for (int i = t0; i < NH * NT * NT; i += stride) {
    int h = i >> 12, ij = i & 4095;
    biasb[i] = bias_table[rel_index[ij] * NH + h];
  }
}

// ------------------------------------------------------------ qkv GEMM -----
// QKV = x[131072,512] @ qkv_w[512,1536]; scatter to q/k/vT f16 buffers.
// Double-buffered LDS, one barrier per K-step; scalar (per-block) q/k/v split.
__global__ __launch_bounds__(256) void qkv_gemm_kernel(
    const float* __restrict__ x, const h16* __restrict__ wT,
    h16* __restrict__ qb, h16* __restrict__ kb, h16* __restrict__ vb) {
  __shared__ h16 lA[2][128 * 32];
  __shared__ h16 lB[2][128 * 32];
  const int tid = threadIdx.x;
  const int lane = tid & 31, w = tid >> 5;
  const int lo = lane & 15, hi = lane >> 4;
  const int mw = w & 3, nw = w >> 2;                // 4x2 wave grid -> 32x64 per wave
  const int m0 = blockIdx.y * 128, n0 = blockIdx.x * 128;
  const int lrow = tid >> 1, lcol = (tid & 1) * 16;

  const float* asrc = x + (size_t)(m0 + lrow) * CH + lcol;
  const h16*   bsrc = wT + (size_t)(n0 + lrow) * CH + lcol;

  v4f fa0, fa1, fa2, fa3;
  v16h gb;
  auto gload = [&](int kt) {
    const float* s = asrc + kt * 32;
    fa0 = *(const v4f*)(s);
    fa1 = *(const v4f*)(s + 4);
    fa2 = *(const v4f*)(s + 8);
    fa3 = *(const v4f*)(s + 12);
    gb = *(const v16h*)(bsrc + kt * 32);
  };
  auto lstore = [&](int buf) {
    h16* da = lA[buf] + lrow * 32 + lcol;
#pragma unroll
    for (int i = 0; i < 4; ++i) {
      da[i]      = (h16)fa0[i];
      da[4 + i]  = (h16)fa1[i];
      da[8 + i]  = (h16)fa2[i];
      da[12 + i] = (h16)fa3[i];
    }
    *(v16h*)(lB[buf] + lrow * 32 + lcol) = gb;
  };

  v8f acc[2][4] = {};
  gload(0);
  lstore(0);
  __syncthreads();
  for (int kt = 0; kt < 16; ++kt) {
    const int cur = kt & 1;
    if (kt < 15) gload(kt + 1);          // prefetch next tile under the WMMAs
    v16h av[2], bv[4];
#pragma unroll
    for (int mi = 0; mi < 2; ++mi)
      av[mi] = load_a_frag(lA[cur], 32, mw * 32 + mi * 16 + lo, 0, hi);
#pragma unroll
    for (int ni = 0; ni < 4; ++ni)
      bv[ni] = load_b_frag(lB[cur], 32, nw * 64 + ni * 16 + lo, 0, hi);
#pragma unroll
    for (int mi = 0; mi < 2; ++mi)
#pragma unroll
      for (int ni = 0; ni < 4; ++ni)
        acc[mi][ni] = wmma_f16(av[mi], bv[ni], acc[mi][ni]);
    if (kt < 15) lstore(cur ^ 1);
    __syncthreads();
  }

  // scatter: sec is uniform per block (n0 multiple of 128; 512 % 128 == 0)
  const int sec = n0 >> 9;               // 0 -> q, 1 -> k, 2 -> v
  if (sec == 2) {
#pragma unroll
    for (int mi = 0; mi < 2; ++mi) {
#pragma unroll
      for (int ni = 0; ni < 4; ++ni) {
        const int cc = (n0 & 511) + nw * 64 + ni * 16;  // lane-invariant
        const int h = cc >> 5, dd0 = cc & 31;
        const int mbase = m0 + mw * 32 + mi * 16 + 8 * hi;
        const int b = mbase >> 6, r0 = mbase & 63;
        h16* dst = vb + ((size_t)b * NH + h) * (NT * HD) +
                   (size_t)(dd0 + lo) * NT + r0;
#pragma unroll
        for (int v = 0; v < 8; ++v) dst[v] = (h16)acc[mi][ni][v];
      }
    }
  } else {
    h16* qk = (sec == 0) ? qb : kb;
    const float sc = (sec == 0) ? SCALE : 1.0f;
#pragma unroll
    for (int mi = 0; mi < 2; ++mi) {
#pragma unroll
      for (int ni = 0; ni < 4; ++ni) {
        const int cc = (n0 & 511) + nw * 64 + ni * 16;
        const int h = cc >> 5, dd0 = cc & 31;
        const int mbase = m0 + mw * 32 + mi * 16 + 8 * hi;
        const int b = mbase >> 6, r0 = mbase & 63;
        h16* dst = qk + ((size_t)b * NH + h) * (NT * HD) +
                   (size_t)r0 * HD + dd0 + lo;
#pragma unroll
        for (int v = 0; v < 8; ++v) dst[(size_t)v * HD] = (h16)(acc[mi][ni][v] * sc);
      }
    }
  }
}

// ----------------------------------------------------------- attention -----
// One wave per (b,h): S=q*k^T (+bias in C operand), softmax, attn_map out, O=P*V.
__global__ __launch_bounds__(128) void attn_kernel(
    const h16* __restrict__ qb, const h16* __restrict__ kb,
    const h16* __restrict__ vb, const float* __restrict__ biasb,
    float* __restrict__ attn_map, h16* __restrict__ ao) {
  __shared__ h16 pbuf[4][64 * 80];   // P staged per wave, ld=80 keeps 32B alignment
  const int tid = threadIdx.x;
  const int lane = tid & 31, w = tid >> 5;
  const int lo = lane & 15, hi = lane >> 4;
  const int b = blockIdx.x;
  const int h = blockIdx.y * 4 + w;
  const h16* qh = qb + ((size_t)b * NH + h) * (NT * HD);
  const h16* kh = kb + ((size_t)b * NH + h) * (NT * HD);
  const h16* vh = vb + ((size_t)b * NH + h) * (NT * HD);
  const float* bh = biasb + h * (NT * NT);

  v16h kf[4];
#pragma unroll
  for (int nb = 0; nb < 4; ++nb) kf[nb] = load_b_frag(kh, HD, nb * 16 + lo, 0, hi);

  v8f s[4][4];
#pragma unroll
  for (int mb = 0; mb < 4; ++mb) {
    v16h qa = load_a_frag(qh, HD, mb * 16 + lo, 0, hi);
#pragma unroll
    for (int nb = 0; nb < 4; ++nb) {
      v8f c;
#pragma unroll
      for (int v = 0; v < 8; ++v)
        c[v] = bh[(mb * 16 + v + 8 * hi) * NT + nb * 16 + lo];
      s[mb][nb] = wmma_f16(qa, kf[nb], c);
    }
  }

  // softmax over rows; row m lives in 16 lanes of one half-wave (xor<=8 stays inside)
  float* am = attn_map + ((size_t)b * NH + h) * (NT * NT);
  h16* pw = pbuf[w];
#pragma unroll
  for (int mb = 0; mb < 4; ++mb) {
#pragma unroll
    for (int v = 0; v < 8; ++v) {
      float mx = fmaxf(fmaxf(s[mb][0][v], s[mb][1][v]),
                       fmaxf(s[mb][2][v], s[mb][3][v]));
#pragma unroll
      for (int off = 1; off < 16; off <<= 1)
        mx = fmaxf(mx, __shfl_xor(mx, off, 32));
      float sum = 0.f;
#pragma unroll
      for (int nb = 0; nb < 4; ++nb) {
        float e = __expf(s[mb][nb][v] - mx);
        s[mb][nb][v] = e;
        sum += e;
      }
#pragma unroll
      for (int off = 1; off < 16; off <<= 1) sum += __shfl_xor(sum, off, 32);
      const float inv = 1.0f / sum;
      const int m = mb * 16 + v + 8 * hi;
#pragma unroll
      for (int nb = 0; nb < 4; ++nb) {
        const float pv = s[mb][nb][v] * inv;
        am[(size_t)m * NT + nb * 16 + lo] = pv;
        pw[m * 80 + nb * 16 + lo] = (h16)pv;
      }
    }
  }
  __syncthreads();

  // O = P[64,64] @ V[64,32] using vT ([32,64] row-major) as B operand
  v16h vf[2][2];
#pragma unroll
  for (int nb = 0; nb < 2; ++nb)
#pragma unroll
    for (int kc = 0; kc < 2; ++kc)
      vf[nb][kc] = load_b_frag(vh, NT, nb * 16 + lo, kc * 32, hi);

  h16* aoh = ao + (size_t)b * NT * CH + h * HD;
#pragma unroll
  for (int mb = 0; mb < 4; ++mb) {
    v16h pa0 = load_a_frag(pw, 80, mb * 16 + lo, 0, hi);
    v16h pa1 = load_a_frag(pw, 80, mb * 16 + lo, 32, hi);
#pragma unroll
    for (int nb = 0; nb < 2; ++nb) {
      v8f o = {};
      o = wmma_f16(pa0, vf[nb][0], o);
      o = wmma_f16(pa1, vf[nb][1], o);
#pragma unroll
      for (int v = 0; v < 8; ++v)
        aoh[(size_t)(mb * 16 + v + 8 * hi) * CH + nb * 16 + lo] = (h16)o[v];
    }
  }
}

// ----------------------------------------------------------- proj GEMM -----
// out = ao[131072,512] @ proj_w[512,512] + proj_b  (fp32 out), double-buffered.
__global__ __launch_bounds__(256) void proj_gemm_kernel(
    const h16* __restrict__ ao, const h16* __restrict__ pT,
    const float* __restrict__ pb, float* __restrict__ out) {
  __shared__ h16 lA[2][128 * 32];
  __shared__ h16 lB[2][128 * 32];
  const int tid = threadIdx.x;
  const int lane = tid & 31, w = tid >> 5;
  const int lo = lane & 15, hi = lane >> 4;
  const int mw = w & 3, nw = w >> 2;
  const int m0 = blockIdx.y * 128, n0 = blockIdx.x * 128;
  const int lrow = tid >> 1, lcol = (tid & 1) * 16;

  const h16* asrc = ao + (size_t)(m0 + lrow) * CH + lcol;
  const h16* bsrc = pT + (size_t)(n0 + lrow) * CH + lcol;

  v16h ga, gb;
  auto gload = [&](int kt) {
    ga = *(const v16h*)(asrc + kt * 32);
    gb = *(const v16h*)(bsrc + kt * 32);
  };
  auto lstore = [&](int buf) {
    *(v16h*)(lA[buf] + lrow * 32 + lcol) = ga;
    *(v16h*)(lB[buf] + lrow * 32 + lcol) = gb;
  };

  v8f acc[2][4] = {};
  gload(0);
  lstore(0);
  __syncthreads();
  for (int kt = 0; kt < 16; ++kt) {
    const int cur = kt & 1;
    if (kt < 15) gload(kt + 1);
    v16h av[2], bv[4];
#pragma unroll
    for (int mi = 0; mi < 2; ++mi)
      av[mi] = load_a_frag(lA[cur], 32, mw * 32 + mi * 16 + lo, 0, hi);
#pragma unroll
    for (int ni = 0; ni < 4; ++ni)
      bv[ni] = load_b_frag(lB[cur], 32, nw * 64 + ni * 16 + lo, 0, hi);
#pragma unroll
    for (int mi = 0; mi < 2; ++mi)
#pragma unroll
      for (int ni = 0; ni < 4; ++ni)
        acc[mi][ni] = wmma_f16(av[mi], bv[ni], acc[mi][ni]);
    if (kt < 15) lstore(cur ^ 1);
    __syncthreads();
  }
#pragma unroll
  for (int mi = 0; mi < 2; ++mi) {
#pragma unroll
    for (int ni = 0; ni < 4; ++ni) {
      const int n = n0 + nw * 64 + ni * 16 + lo;
      const int mbase = m0 + mw * 32 + mi * 16 + 8 * hi;
      const float bias = pb[n];
#pragma unroll
      for (int v = 0; v < 8; ++v)
        out[(size_t)(mbase + v) * CH + n] = acc[mi][ni][v] + bias;
    }
  }
}

// --------------------------------------------------------------- launch ----
extern "C" void kernel_launch(void* const* d_in, const int* in_sizes, int n_in,
                              void* d_out, int out_size, void* d_ws, size_t ws_size,
                              hipStream_t stream) {
  const float* x          = (const float*)d_in[0];
  const float* qkv_w      = (const float*)d_in[1];
  const float* proj_w     = (const float*)d_in[2];
  const float* proj_b     = (const float*)d_in[3];
  const float* bias_table = (const float*)d_in[4];
  const int*   rel_index  = (const int*)d_in[5];

  float* out      = (float*)d_out;
  float* attn_map = out + (size_t)NW * NT * CH;  // tuple output, concatenated

  char* p = (char*)d_ws;
  auto alloc = [&](size_t bytes) -> void* {
    void* r = (void*)p;
    p += (bytes + 255) & ~(size_t)255;
    return r;
  };
  h16*   qkvT  = (h16*)alloc((size_t)1536 * 512 * sizeof(h16));
  h16*   projT = (h16*)alloc((size_t)512 * 512 * sizeof(h16));
  float* biasb = (float*)alloc((size_t)NH * NT * NT * sizeof(float));
  h16*   qb    = (h16*)alloc((size_t)NW * NH * NT * HD * sizeof(h16));
  h16*   kb    = (h16*)alloc((size_t)NW * NH * NT * HD * sizeof(h16));
  h16*   vb    = (h16*)alloc((size_t)NW * NH * NT * HD * sizeof(h16));
  h16*   ao    = (h16*)alloc((size_t)NW * NT * CH * sizeof(h16));

  prep_kernel<<<512, 256, 0, stream>>>(qkv_w, proj_w, bias_table, rel_index,
                                       qkvT, projT, biasb);
  qkv_gemm_kernel<<<dim3(12, 1024), 256, 0, stream>>>(x, qkvT, qb, kb, vb);
  attn_kernel<<<dim3(NW, 4), 128, 0, stream>>>(qb, kb, vb, biasb, attn_map, ao);
  proj_gemm_kernel<<<dim3(4, 1024), 256, 0, stream>>>(ao, projT, proj_b, out);
}
